// GRU4Rec_4329327034833
// MI455X (gfx1250) — matrix operand order repaired and checked by reading, not compiled, not run on hardware
//
#include <hip/hip_runtime.h>

#define BDIM 256
#define HID 256
#define VOC 50001
#define VOCP 50048      // 391 * 128
#define LSTEPS 32
#define TOPK 100
#define DSDIM 64
#define G3 768
#define TAU 1.0f
#define EPS_LN 1e-8f
#define TIE_MAX 256

typedef __attribute__((ext_vector_type(16))) _Float16 v16h;
typedef __attribute__((ext_vector_type(8)))  _Float16 v8h;
typedef __attribute__((ext_vector_type(8)))  float    v8f;

// ---------------- prep kernels (run every launch; deterministic) ----------------

// item_emb [HID, VOC] f32 -> f16 padded [HID, VOCP], pad = 0
__global__ __launch_bounds__(256) void k_conv_item(const float* __restrict__ w,
                                                   _Float16* __restrict__ o) {
  int i = blockIdx.x * 256 + threadIdx.x;
  if (i >= HID * VOCP) return;
  int k = i / VOCP, v = i - k * VOCP;
  o[i] = (v < VOC) ? (_Float16)w[(size_t)k * VOC + v] : (_Float16)0.f;
}

// W [G3, HID] f32 -> WT [HID, G3] f16  (B-matrix layout: row k contiguous in n)
__global__ __launch_bounds__(256) void k_transpose_g(const float* __restrict__ w,
                                                     _Float16* __restrict__ o) {
  int i = blockIdx.x * 256 + threadIdx.x;
  if (i >= G3 * HID) return;
  int h = i / G3, j = i - h * G3;
  o[i] = (_Float16)w[(size_t)j * HID + h];
}

// X0 = item_emb[:, seqs[b]] (f16), H = H0, seqs -> first B entries of flat seq output
__global__ __launch_bounds__(256) void k_init(const int* __restrict__ seqs,
                                              const float* __restrict__ item,
                                              const float* __restrict__ H0,
                                              _Float16* __restrict__ Xf16,
                                              float* __restrict__ Hf32,
                                              _Float16* __restrict__ Hf16,
                                              float* __restrict__ seqflat) {
  int i = blockIdx.x * 256 + threadIdx.x;   // 65536 threads
  int b = i >> 8, h = i & 255;
  Xf16[i] = (_Float16)item[(size_t)h * VOC + seqs[b]];
  Hf32[i] = H0[i];
  Hf16[i] = (_Float16)H0[i];
  if (i < BDIM) seqflat[i] = (float)seqs[i];
}

// ---------------- WMMA GEMM: C[256, Npad] = A[256,256]f16 @ B[256,Npad]f16 ----------------
// grid = (Npad/128, 256/64), block = 256 (8 waves).
// Each wave: 16 N-cols x 64 M-rows (4 subtiles). Per k-step: one B fragment (2x b128)
// reused by 4 WMMAs; all 4 A fragments preloaded from LDS in one clause so the
// ds-waits are incremental instead of full drains before each WMMA.
__global__ __launch_bounds__(256) void k_gemm(const _Float16* __restrict__ A,
                                              const _Float16* __restrict__ Bm,
                                              float* __restrict__ C, int Npad) {
  __shared__ __align__(32) _Float16 As[64][256];   // 32 KB A tile
  const int m0 = blockIdx.y * 64;
  const int nblk = blockIdx.x * 128;
  const int tid = threadIdx.x;
#pragma unroll
  for (int it = 0; it < 4; ++it) {
    int cidx = tid + 256 * it;                 // 1024 chunks of 16 halfs
    int row = cidx >> 4, chunk = cidx & 15;
    const _Float16* src = A + (size_t)(m0 + row) * HID + chunk * 16;
    *(v8h*)&As[row][chunk * 16]     = *(const v8h*)src;
    *(v8h*)&As[row][chunk * 16 + 8] = *(const v8h*)(src + 8);
  }
  __syncthreads();
  const int lane = tid & 31, wave = tid >> 5;
  const int n0 = nblk + wave * 16;
  const int mrow = lane & 15, hi = lane >> 4;

  v8f acc[4];
#pragma unroll
  for (int s = 0; s < 4; ++s) acc[s] = (v8f){};

  auto mk_a = [&](int msub, int k0) -> v16h {
    // ISA 16-bit A 16x32 layout: two 8-half chunks at K-offsets {8*hi, 16+8*hi}
    v8h a0 = *(const v8h*)&As[msub * 16 + mrow][k0 + 8 * hi];
    v8h a1 = *(const v8h*)&As[msub * 16 + mrow][k0 + 16 + 8 * hi];
    v16h a;
#pragma unroll
    for (int j = 0; j < 8; ++j) { a[j] = a0[j]; a[j + 8] = a1[j]; }
    return a;
  };

#pragma unroll
  for (int t = 0; t < 8; ++t) {
    const int k0 = t * 32;
    // B fragment: lane holds column n, K-halfs at K-offsets {8*hi, 16+8*hi}
    const int nb = n0 + (lane & 15);
    const _Float16* bp = Bm + nb;
    v16h bb;
#pragma unroll
    for (int j = 0; j < 8; ++j) {
      bb[j] = bp[(size_t)(k0 + 8 * hi + j) * Npad];
      bb[j + 8] = bp[(size_t)(k0 + 16 + 8 * hi + j) * Npad];
    }
    // preload all 4 A fragments (distinct registers -> ds_loads clause together)
    v16h a0 = mk_a(0, k0);
    v16h a1 = mk_a(1, k0);
    v16h a2 = mk_a(2, k0);
    v16h a3 = mk_a(3, k0);
    acc[0] = __builtin_amdgcn_wmma_f32_16x16x32_f16(false, a0, false, bb, (short)0, acc[0], false, false);
    acc[1] = __builtin_amdgcn_wmma_f32_16x16x32_f16(false, a1, false, bb, (short)0, acc[1], false, false);
    acc[2] = __builtin_amdgcn_wmma_f32_16x16x32_f16(false, a2, false, bb, (short)0, acc[2], false, false);
    acc[3] = __builtin_amdgcn_wmma_f32_16x16x32_f16(false, a3, false, bb, (short)0, acc[3], false, false);
  }
  const int ncol = n0 + (lane & 15);
#pragma unroll
  for (int s = 0; s < 4; ++s) {
    float* crow = C + (size_t)(m0 + s * 16 + 8 * hi) * Npad + ncol;
#pragma unroll
    for (int r = 0; r < 8; ++r) crow[(size_t)r * Npad] = acc[s][r];
  }
}

// ---------------- GRU gates + layernorm (block per b, thread per h) ----------------
__global__ __launch_bounds__(256) void k_gates(const float* __restrict__ gi,
                                               const float* __restrict__ gh,
                                               const float* __restrict__ bih,
                                               const float* __restrict__ bhh,
                                               const float* __restrict__ ln_g,
                                               const float* __restrict__ ln_b,
                                               float* __restrict__ Hf32,
                                               _Float16* __restrict__ Hf16,
                                               _Float16* __restrict__ Araw) {
  const int b = blockIdx.x, h = threadIdx.x;
  __shared__ float red[256];
  __shared__ float s_mean, s_var;
  const float* gib = gi + (size_t)b * G3;
  const float* ghb = gh + (size_t)b * G3;
  float ir = gib[h]           + bih[h],           hr = ghb[h]           + bhh[h];
  float iz = gib[h + HID]     + bih[h + HID],     hz = ghb[h + HID]     + bhh[h + HID];
  float in_= gib[h + 2 * HID] + bih[h + 2 * HID], hn = ghb[h + 2 * HID] + bhh[h + 2 * HID];
  float r = 1.f / (1.f + expf(-(ir + hr)));
  float z = 1.f / (1.f + expf(-(iz + hz)));
  float n = tanhf(in_ + r * hn);
  float Hp = Hf32[b * HID + h];
  float hraw = (1.f - z) * n + z * Hp;
  red[h] = hraw; __syncthreads();
  for (int off = 128; off > 0; off >>= 1) { if (h < off) red[h] += red[h + off]; __syncthreads(); }
  if (h == 0) s_mean = red[0] * (1.f / HID);
  __syncthreads();
  float d = hraw - s_mean;
  red[h] = d * d; __syncthreads();
  for (int off = 128; off > 0; off >>= 1) { if (h < off) red[h] += red[h + off]; __syncthreads(); }
  if (h == 0) s_var = red[0] * (1.f / HID);
  __syncthreads();
  float hln = d * rsqrtf(s_var + EPS_LN) * ln_g[h] + ln_b[h];
  Hf32[b * HID + h] = hln;
  Hf16[b * HID + h] = (_Float16)hln;
  Araw[b * HID + h] = (_Float16)hraw;   // X for logits GEMM (pre-LN GRU output)
}

// ---------------- exact top-K (sorted desc) via radix-select, block per row ----------------
__device__ __forceinline__ unsigned monokey(unsigned b) {
  return (b & 0x80000000u) ? ~b : (b | 0x80000000u);
}
__device__ __forceinline__ float unmono(unsigned k) {
  unsigned b = (k & 0x80000000u) ? (k ^ 0x80000000u) : ~k;
  return __uint_as_float(b);
}

__global__ __launch_bounds__(256) void k_topk(const float* __restrict__ logits,
                                              float* __restrict__ vals,
                                              int* __restrict__ idxs) {
  const int b = blockIdx.x, tid = threadIdx.x;
  const float* row = logits + (size_t)b * VOCP;
  __shared__ unsigned hist[256];
  __shared__ unsigned sh_prefix, sh_kneed, cntG, cntT;
  __shared__ unsigned long long comp[128];
  __shared__ int tiebuf[TIE_MAX];
  if (tid == 0) { sh_prefix = 0u; sh_kneed = TOPK; cntG = 0u; cntT = 0u; }
  unsigned prefixMask = 0u;
  for (int pass = 0; pass < 4; ++pass) {
    const int shift = 24 - 8 * pass;
    hist[tid] = 0u;
    __syncthreads();
    const unsigned prefix = sh_prefix;
    for (int v = tid; v < VOC; v += 256) {
      unsigned key = monokey(__float_as_uint(row[v]));
      if ((key & prefixMask) == prefix) atomicAdd(&hist[(key >> shift) & 255u], 1u);
    }
    __syncthreads();
    if (tid == 0) {
      unsigned cum = 0u; int sel = 0;
      for (int bin = 255; bin >= 0; --bin) {
        unsigned hh = hist[bin];
        if (cum + hh >= sh_kneed) { sel = bin; break; }
        cum += hh;
      }
      sh_prefix = prefix | ((unsigned)sel << shift);
      sh_kneed -= cum;
    }
    __syncthreads();
    prefixMask |= (0xFFu << shift);
  }
  const unsigned T = sh_prefix;      // key of K-th largest
  const unsigned needT = sh_kneed;   // how many ties at T to take
  for (int v = tid; v < VOC; v += 256) {
    unsigned key = monokey(__float_as_uint(row[v]));
    if (key > T) {
      unsigned p = atomicAdd(&cntG, 1u);
      if (p < TOPK) comp[p] = ((unsigned long long)key << 32) | (unsigned)(~(unsigned)v);
    } else if (key == T) {
      unsigned p = atomicAdd(&cntT, 1u);
      if (p < TIE_MAX) tiebuf[p] = v;
    }
  }
  __syncthreads();
  if (tid >= TOPK && tid < 128) comp[tid] = 0ull;   // pad sorts to the end
  if (tid == 0) {
    int nt = (int)(cntT < TIE_MAX ? cntT : TIE_MAX);
    for (int i = 1; i < nt; ++i) {                  // ties: smallest index first (jax)
      int x = tiebuf[i]; int j = i - 1;
      while (j >= 0 && tiebuf[j] > x) { tiebuf[j + 1] = tiebuf[j]; --j; }
      tiebuf[j + 1] = x;
    }
    unsigned base = TOPK - needT;                   // == #strictly-greater
    for (unsigned t = 0; t < needT && (int)t < nt; ++t)
      comp[base + t] = ((unsigned long long)T << 32) | (unsigned)(~(unsigned)tiebuf[t]);
  }
  __syncthreads();
  // bitonic sort 128 elements, descending (key desc; ~idx => idx asc on ties)
  for (int size = 2; size <= 128; size <<= 1) {
    for (int stride = size >> 1; stride > 0; stride >>= 1) {
      if (tid < 128) {
        int partner = tid ^ stride;
        if (partner > tid) {
          unsigned long long a = comp[tid], c2 = comp[partner];
          bool doSwap = ((tid & size) == 0) ? (a < c2) : (a > c2);
          if (doSwap) { comp[tid] = c2; comp[partner] = a; }
        }
      }
      __syncthreads();
    }
  }
  if (tid < TOPK) {
    unsigned long long e = comp[tid];
    vals[b * TOPK + tid] = unmono((unsigned)(e >> 32));
    idxs[b * TOPK + tid] = (int)(~(unsigned)(e & 0xFFFFFFFFu));
  }
}

// ---------------- gumbel argmax + column gathers (block per row) ----------------
__global__ __launch_bounds__(256) void k_pick(const float* __restrict__ vals,
                                              const int* __restrict__ idxs,
                                              const float* __restrict__ gum,
                                              const float* __restrict__ Temb,
                                              const float* __restrict__ Semb,
                                              const float* __restrict__ item,
                                              int l,
                                              float* __restrict__ t_out,
                                              float* __restrict__ s_out,
                                              float* __restrict__ seqflat,
                                              _Float16* __restrict__ Xf16) {
  const int b = blockIdx.x, tid = threadIdx.x;
  __shared__ float ssc[128];
  __shared__ int sk[128];
  __shared__ int s_v;
  if (tid < 128) {
    float s = -3.4e38f;
    if (tid < TOPK) {
      float u = gum[((size_t)l * BDIM + b) * TOPK + tid];
      float g = -logf(-logf(u));
      s = (vals[b * TOPK + tid] + g) * (1.0f / TAU);
    }
    ssc[tid] = s; sk[tid] = tid;
  }
  __syncthreads();
  for (int off = 64; off > 0; off >>= 1) {
    if (tid < off) {
      float a = ssc[tid], c2 = ssc[tid + off];
      int ka = sk[tid], kc = sk[tid + off];
      if (c2 > a || (c2 == a && kc < ka)) { ssc[tid] = c2; sk[tid] = kc; }
    }
    __syncthreads();
  }
  if (tid == 0) s_v = idxs[b * TOPK + sk[0]];
  __syncthreads();
  const int v = s_v;
  t_out[((size_t)b * LSTEPS + l) * HID + tid] = Temb[(size_t)tid * VOC + v];
  Xf16[b * HID + tid] = (_Float16)item[(size_t)tid * VOC + v];
  if (tid < DSDIM) s_out[((size_t)b * LSTEPS + l) * DSDIM + tid] = Semb[(size_t)tid * VOC + v];
  if (tid == 0) seqflat[BDIM + l * BDIM + b] = (float)v;
}

// ---------------- host ----------------
extern "C" void kernel_launch(void* const* d_in, const int* in_sizes, int n_in,
                              void* d_out, int out_size, void* d_ws, size_t ws_size,
                              hipStream_t stream) {
  (void)in_sizes; (void)n_in; (void)out_size; (void)ws_size;
  const int*   seqs = (const int*)d_in[0];
  const float* item = (const float*)d_in[1];
  const float* Wih  = (const float*)d_in[2];
  const float* Whh  = (const float*)d_in[3];
  const float* bih  = (const float*)d_in[4];
  const float* bhh  = (const float*)d_in[5];
  const float* ln_g = (const float*)d_in[6];
  const float* ln_b = (const float*)d_in[7];
  const float* Temb = (const float*)d_in[8];
  const float* Semb = (const float*)d_in[9];
  const float* H0   = (const float*)d_in[10];
  const float* gum  = (const float*)d_in[11];

  float* t_out   = (float*)d_out;
  float* s_out   = t_out + (size_t)BDIM * LSTEPS * HID;
  float* seqflat = s_out + (size_t)BDIM * LSTEPS * DSDIM;

  char* ws = (char*)d_ws;
  size_t cur = 0;
  auto alloc = [&](size_t bytes) -> void* {
    void* p = ws + cur;
    cur = (cur + bytes + 511) & ~(size_t)511;
    return p;
  };
  _Float16* Wf16   = (_Float16*)alloc((size_t)HID * VOCP * 2); // item emb f16, padded
  float*    logitsP= (float*)   alloc((size_t)BDIM * VOCP * 4);
  _Float16* WihT   = (_Float16*)alloc((size_t)HID * G3 * 2);
  _Float16* WhhT   = (_Float16*)alloc((size_t)HID * G3 * 2);
  float*    gi     = (float*)   alloc((size_t)BDIM * G3 * 4);
  float*    gh     = (float*)   alloc((size_t)BDIM * G3 * 4);
  _Float16* Xf16   = (_Float16*)alloc((size_t)BDIM * HID * 2);
  _Float16* Hf16   = (_Float16*)alloc((size_t)BDIM * HID * 2);
  _Float16* Araw   = (_Float16*)alloc((size_t)BDIM * HID * 2);
  float*    Hf32   = (float*)   alloc((size_t)BDIM * HID * 4);
  float*    vals   = (float*)   alloc((size_t)BDIM * TOPK * 4);
  int*      idxs   = (int*)     alloc((size_t)BDIM * TOPK * 4);

  // prep (constant work, redone each call for determinism)
  k_conv_item<<<(HID * VOCP + 255) / 256, 256, 0, stream>>>(item, Wf16);
  k_transpose_g<<<(G3 * HID + 255) / 256, 256, 0, stream>>>(Wih, WihT);
  k_transpose_g<<<(G3 * HID + 255) / 256, 256, 0, stream>>>(Whh, WhhT);
  k_init<<<BDIM, 256, 0, stream>>>(seqs, item, H0, Xf16, Hf32, Hf16, seqflat);

  for (int l = 0; l < LSTEPS; ++l) {
    k_gemm<<<dim3(G3 / 128, BDIM / 64), 256, 0, stream>>>(Xf16, WihT, gi, G3);
    k_gemm<<<dim3(G3 / 128, BDIM / 64), 256, 0, stream>>>(Hf16, WhhT, gh, G3);
    k_gates<<<BDIM, 256, 0, stream>>>(gi, gh, bih, bhh, ln_g, ln_b, Hf32, Hf16, Araw);
    k_gemm<<<dim3(VOCP / 128, BDIM / 64), 256, 0, stream>>>(Araw, Wf16, logitsP, VOCP);
    k_topk<<<BDIM, 256, 0, stream>>>(logitsP, vals, idxs);
    k_pick<<<BDIM, 256, 0, stream>>>(vals, idxs, gum, Temb, Semb, item, l,
                                     t_out, s_out, seqflat, Xf16);
  }
}
